// GeodesicAwareAttention_21002390078157
// MI455X (gfx1250) — matrix operand order repaired and checked by reading, not compiled, not run on hardware
//
#include <hip/hip_runtime.h>
#include <hip/hip_bf16.h>

// Problem constants (from reference)
constexpr int kB  = 2;
constexpr int kC  = 256;
constexpr int kN  = 4096;   // H*W = 64*64
constexpr int kNH = 4;
constexpr int kHD = 64;

typedef _Float16 f16;
typedef __attribute__((ext_vector_type(16))) _Float16 v16h;
typedef __attribute__((ext_vector_type(8)))  _Float16 v8h;
typedef __attribute__((ext_vector_type(4)))  _Float16 v4h;
typedef __attribute__((ext_vector_type(8)))  float    v8f;

static __device__ __forceinline__ v8f wmma16x16x32(v16h a, v16h b, v8f c) {
  // D = A(16x32 f16) * B(32x16 f16) + C(16x16 f32)
  return __builtin_amdgcn_wmma_f32_16x16x32_f16(false, a, false, b, (short)0, c,
                                                false, false);
}

static __device__ __forceinline__ v16h cat8(v8h lo, v8h hi) {
  return __builtin_shufflevector(lo, hi, 0, 1, 2, 3, 4, 5, 6, 7, 8, 9, 10, 11,
                                 12, 13, 14, 15);
}

// ---------------------------------------------------------------------------
// Kernel 0: one-time weight conversion f32 -> f16 (wq scaled by hd^-0.5).
// wh layout: [4][C*C] = wq*0.125 | wk | wv | wo. ~1.5 MB total traffic.
// ---------------------------------------------------------------------------
__global__ __launch_bounds__(256) void cvt_weights(
    const float* __restrict__ wq, const float* __restrict__ wk,
    const float* __restrict__ wv, const float* __restrict__ wo,
    f16* __restrict__ wh) {
  const int m = blockIdx.y;
  const float* src = (m == 0) ? wq : (m == 1) ? wk : (m == 2) ? wv : wo;
  const float sc = (m == 0) ? 0.125f : 1.0f;
  const int off = (blockIdx.x * 256 + threadIdx.x) * 4;
  const float4 v = *(const float4*)(src + off);
  v4h o = {(f16)(v.x * sc), (f16)(v.y * sc), (f16)(v.z * sc), (f16)(v.w * sc)};
  *(v4h*)(wh + (size_t)m * kC * kC + off) = o;
}

// ---------------------------------------------------------------------------
// Kernel 1: fused Q/K/V projection, f16 weights loaded directly in B layout.
//   Q[b,n,j] = (sum_c x[b,c,n] * wq[j,c]) * hd^-0.5 + bq[j]*hd^-0.5
//   Q,K stored [B,N,C] f16; V stored transposed per head: Vt[b,h,d,n] f16.
// Block: 128 threads (4 waves) per (b, 16-row n-tile); waves split the 48
// (matrix, out-col-tile) jobs. A-operands staged via LDS once, kept in regs.
// ---------------------------------------------------------------------------
__global__ __launch_bounds__(128) void qkv_proj(
    const float* __restrict__ x, const f16* __restrict__ wh,
    const float* __restrict__ bq, const float* __restrict__ bk,
    const float* __restrict__ bvb,
    f16* __restrict__ Qh, f16* __restrict__ Kh, f16* __restrict__ Vt) {
  __shared__ float xs[kC * 16];  // x_flat strip: [c][n_local], 16 KB
  const int b     = blockIdx.x / (kN / 16);
  const int nbase = (blockIdx.x % (kN / 16)) * 16;
  const int tid   = threadIdx.x;
  for (int i = tid; i < kC * 16; i += 128) {
    int c = i >> 4, nl = i & 15;
    xs[i] = x[((size_t)b * kC + c) * kN + nbase + nl];
  }
  __syncthreads();
  const int wave = tid >> 5, lane = tid & 31;
  const int l16 = lane & 15, hiw = lane >> 4;

  // Preload all 8 A tiles (x_flat [16n x 32c]) in WMMA A layout (shared by jobs)
  v16h a[8];
#pragma unroll
  for (int ks = 0; ks < 8; ++ks) {
    int cb = ks * 32 + hiw * 16;
#pragma unroll
    for (int t = 0; t < 8; ++t) {
      a[ks][t]     = (f16)xs[(cb + t) * 16 + l16];
      a[ks][t + 8] = (f16)xs[(cb + 8 + t) * 16 + l16];
    }
  }

  for (int job = wave; job < 48; job += 4) {
    const int mat = job >> 4;        // 0=Q, 1=K, 2=V
    const int jt  = job & 15;        // out-column tile
    const f16* wmat = wh + (size_t)mat * kC * kC;
    const float* bias = (mat == 0) ? bq : (mat == 1) ? bk : bvb;
    const int j = jt * 16 + l16;     // out column handled by this lane (B-layout N)
    v8f acc = {};
    for (int ks = 0; ks < 8; ++ks) {
      // B layout: lane = out-col, K = hiw*16 + 0..15 -> one contiguous 32B run
      v16h bb = *(const v16h*)(wmat + (size_t)j * kC + ks * 32 + hiw * 16);
      acc = wmma16x16x32(a[ks], bb, acc);
    }
    const float bj = bias[j] * ((mat == 0) ? 0.125f : 1.0f);
    if (mat < 2) {
      f16* dst = (mat == 0) ? Qh : Kh;
#pragma unroll
      for (int r = 0; r < 8; ++r) {
        int n = nbase + hiw * 8 + r;  // C-layout: row m = hiw*8 + r
        dst[((size_t)b * kN + n) * kC + j] = (f16)(acc[r] + bj);
      }
    } else {
      // V transposed per head: Vt[b, h, d, n]; rows of acc are contiguous n.
      int h = j >> 6, d = j & 63;
      v8h pk;
#pragma unroll
      for (int r = 0; r < 8; ++r) pk[r] = (f16)(acc[r] + bj);
      *(v8h*)(Vt + (((size_t)b * kNH + h) * kHD + d) * kN + nbase + hiw * 8) = pk;
    }
  }
}

// ---------------------------------------------------------------------------
// Kernel 2: flash attention, shift-free softmax (logits are O(1) by
// construction, so exp() cannot overflow and softmax is shift-invariant).
// One wave per 16-query tile; 4 waves/block. Per 32-key step:
//   4x WMMA (S = QK^T, scale pre-folded into Q), mask->exp->P,
//   per-lane partial row sums (no cross-lane traffic in the loop),
//   C->A relayout of P via wave-local LDS (vectorized b128 re-read),
//   4x WMMA (O += P*V). Row-sum reduction + normalize once at the end.
// ---------------------------------------------------------------------------
__global__ __launch_bounds__(128) void flash_attn(
    const f16* __restrict__ Qh, const f16* __restrict__ Kh,
    const f16* __restrict__ Vt, const unsigned char* __restrict__ mask,
    f16* __restrict__ Oa) {
  __shared__ f16 pbuf[4][16][32];  // per-wave P staging (wave-local, no barrier)
  const int wave = threadIdx.x >> 5, lane = threadIdx.x & 31;
  const int l16 = lane & 15, hiw = lane >> 4;
  const int qtile = blockIdx.x * 4 + wave;
  const int h = blockIdx.y, b = blockIdx.z;
  const int qbase = qtile * 16;

  // Q tile A-operands (two 16x32 slabs covering head_dim 64): A-layout slots
  // 0..7 / 8..15 are two contiguous 16B runs -> vector loads.
  const f16* qp = Qh + ((size_t)b * kN + qbase + l16) * kC + h * kHD;
  v16h qa[2];
#pragma unroll
  for (int s = 0; s < 2; ++s)
    qa[s] = cat8(*(const v8h*)(qp + s * 32 + hiw * 16),
                 *(const v8h*)(qp + s * 32 + hiw * 16 + 8));

  v8f o0 = {}, o1 = {}, o2 = {}, o3 = {};
  float rsum[8];
#pragma unroll
  for (int r = 0; r < 8; ++r) rsum[r] = 0.0f;

  const f16* kbase = Kh + ((size_t)b * kN + l16) * kC + h * kHD + hiw * 16;
  const f16* vbase =
      Vt + (((size_t)b * kNH + h) * kHD + l16) * kN + hiw * 16;
  const unsigned char* mbase = mask + ((size_t)b * kN + qbase + hiw * 8) * kN;

  for (int kv = 0; kv < kN; kv += 32) {
    // S = Q K^T : 2 col-tiles x 2 K-dim steps
    v8f s0 = {}, s1 = {};
#pragma unroll
    for (int s = 0; s < 2; ++s) {
      const f16* kp = kbase + (size_t)kv * kC + s * 32;
      s0 = wmma16x16x32(qa[s], *(const v16h*)kp, s0);
      s1 = wmma16x16x32(qa[s], *(const v16h*)(kp + 16 * kC), s1);
    }
    // mask -> P = exp(S) (shift-free), per-lane partial row sums
    const unsigned char* mrow = mbase + kv;
    __builtin_prefetch(mrow + 128, 0, 0);  // global_prefetch_b8 on mask stream
    float p[16];
#pragma unroll
    for (int r = 0; r < 8; ++r) {
      float e0 = __expf(s0[r]);
      float e1 = __expf(s1[r]);
      p[r]     = mrow[(size_t)r * kN + l16]      ? e0 : 0.0f;
      p[r + 8] = mrow[(size_t)r * kN + 16 + l16] ? e1 : 0.0f;
      rsum[r] += p[r] + p[r + 8];
    }
    // P: C layout -> LDS (scalar f16 stores, scattered by layout)
#pragma unroll
    for (int r = 0; r < 8; ++r) {
      pbuf[wave][hiw * 8 + r][l16]      = (f16)p[r];
      pbuf[wave][hiw * 8 + r][16 + l16] = (f16)p[r + 8];
    }
    // A-layout re-read: two contiguous 16B runs per lane -> 2x ds_load_b128
    v16h pa = cat8(*(const v8h*)&pbuf[wave][l16][hiw * 16],
                   *(const v8h*)&pbuf[wave][l16][hiw * 16 + 8]);
    // O += P * V  (Vt[b,h,d,n]: lane = d col, contiguous K along n)
    const f16* vp = vbase + kv;
    o0 = wmma16x16x32(pa, *(const v16h*)(vp),           o0);
    o1 = wmma16x16x32(pa, *(const v16h*)(vp + 16 * kN), o1);
    o2 = wmma16x16x32(pa, *(const v16h*)(vp + 32 * kN), o2);
    o3 = wmma16x16x32(pa, *(const v16h*)(vp + 48 * kN), o3);
  }
  // Final row-sum reduction across the 16 lanes of each half, then normalize.
#pragma unroll
  for (int r = 0; r < 8; ++r) {
    float ps = rsum[r];
#pragma unroll
    for (int off = 1; off < 16; off <<= 1) ps += __shfl_xor(ps, off, 32);
    float inv = 1.0f / ps;
    int n = qbase + hiw * 8 + r;
    f16* dst = Oa + ((size_t)b * kN + n) * kC + h * kHD + l16;
    dst[0]  = (f16)(o0[r] * inv);
    dst[16] = (f16)(o1[r] * inv);
    dst[32] = (f16)(o2[r] * inv);
    dst[48] = (f16)(o3[r] * inv);
  }
}

// ---------------------------------------------------------------------------
// Kernel 3: output projection + transpose back to [B,C,N] f32.
//   out[b,j,n] = sum_c Oa[b,n,c]*wo[j,c] + bo[j]
// C-layout rows map to contiguous n -> contiguous f32 stores.
// ---------------------------------------------------------------------------
__global__ __launch_bounds__(128) void out_proj(
    const f16* __restrict__ Oa, const f16* __restrict__ woh,
    const float* __restrict__ bo, float* __restrict__ out) {
  const int b     = blockIdx.x / (kN / 16);
  const int nbase = (blockIdx.x % (kN / 16)) * 16;
  const int wave = threadIdx.x >> 5, lane = threadIdx.x & 31;
  const int l16 = lane & 15, hiw = lane >> 4;

  // Preload A (Oa tile [16n x 256c], row-major f16) in A layout, vector loads
  const f16* ap = Oa + ((size_t)b * kN + nbase + l16) * kC;
  v16h a[8];
#pragma unroll
  for (int ks = 0; ks < 8; ++ks) {
    int cb = ks * 32 + hiw * 16;
    a[ks] = cat8(*(const v8h*)(ap + cb), *(const v8h*)(ap + cb + 8));
  }
  for (int jt = wave; jt < 16; jt += 4) {
    const int j = jt * 16 + l16;
    v8f acc = {};
    for (int ks = 0; ks < 8; ++ks) {
      v16h bb = *(const v16h*)(woh + (size_t)j * kC + ks * 32 + hiw * 16);
      acc = wmma16x16x32(a[ks], bb, acc);
    }
    const float bj = bo[j];
    float* dst = out + ((size_t)b * kC + j) * kN + nbase + hiw * 8;
#pragma unroll
    for (int r = 0; r < 8; ++r) dst[r] = acc[r] + bj;
  }
}

// ---------------------------------------------------------------------------
extern "C" void kernel_launch(void* const* d_in, const int* in_sizes, int n_in,
                              void* d_out, int out_size, void* d_ws,
                              size_t ws_size, hipStream_t stream) {
  (void)in_sizes; (void)n_in; (void)out_size; (void)ws_size;
  const float*         x    = (const float*)d_in[0];
  const unsigned char* mask = (const unsigned char*)d_in[1];  // bool mask
  const float* wq = (const float*)d_in[2];
  const float* bq = (const float*)d_in[3];
  const float* wk = (const float*)d_in[4];
  const float* bk = (const float*)d_in[5];
  const float* wv = (const float*)d_in[6];
  const float* bv = (const float*)d_in[7];
  const float* wo = (const float*)d_in[8];
  const float* bo = (const float*)d_in[9];
  float* out = (float*)d_out;

  const size_t elems = (size_t)kB * kN * kC;  // 2,097,152
  f16* Qh = (f16*)d_ws;          // [B,N,C] f16 (pre-scaled by hd^-0.5)
  f16* Kh = Qh + elems;          // [B,N,C] f16
  f16* Vt = Kh + elems;          // [B,NH,HD,N] f16
  f16* Oa = Vt + elems;          // [B,N,C] f16
  f16* Wh = Oa + elems;          // [4][C*C] f16 weights (total ~16.5 MB ws)

  cvt_weights<<<dim3(kC * kC / (256 * 4), 4), dim3(256), 0, stream>>>(
      wq, wk, wv, wo, Wh);
  qkv_proj<<<dim3(kB * (kN / 16)), dim3(128), 0, stream>>>(
      x, Wh, bq, bk, bv, Qh, Kh, Vt);
  flash_attn<<<dim3(kN / 16 / 4, kNH, kB), dim3(128), 0, stream>>>(
      Qh, Kh, Vt, mask, Oa);
  out_proj<<<dim3(kB * (kN / 16)), dim3(128), 0, stream>>>(
      Oa, Wh + 3 * (size_t)kC * kC, bo, out);
}
